// Iteration_dec_77068893159869
// MI455X (gfx1250) — matrix operand order repaired and checked by reading, not compiled, not run on hardware
//
#include <hip/hip_runtime.h>
#include <hip/hip_bf16.h>

// ---------------------------------------------------------------------------
// MI455X (gfx1250) fused Conv1D(k=5)+BN+ELU x5 stages with permute folding.
// bf16 WMMA 16x16x32, f32 accumulation, all 5 layers of a stage fused in LDS.
// Loop nest keeps B fragments live for exactly one K-iteration (no spills).
// Epilogue stores are unconditional: out-of-range rows hold finite garbage
// that no consumer ever reads (all A-row reads are clamped below Rin).
// ---------------------------------------------------------------------------

typedef __attribute__((ext_vector_type(16))) __bf16 v16bf;
typedef __attribute__((ext_vector_type(8)))  __bf16 v8bf;
typedef __attribute__((ext_vector_type(8)))  float  v8f;

union BFrag16 { v16bf v; v8bf h[2]; };

#define BATCH 256
#define SEQL  2048
#define DIM   50
#define TILE  128           // output positions per workgroup
#define HALO  10            // 5 layers * 2
#define R0    148           // TILE + 2*HALO input rows
#define S1    152           // stage-input LDS stride (150ch + 2 pad); 304B = 16B aligned
#define BUF0_ELEMS (R0 * S1 + 16)   // +16 bf16 zero pad for chunk-4 overhang on last row
#define BUF1_ELEMS (144 * 64)
#define BN_EPS 1e-3f

// Swizzled weight fragment store in workspace: fragment f = 32 lanes x 16 bf16
// at wsB + f*512.  B-matrix (32x16 KxN) register layout assumed (wave32):
//   lane n    (0..15): N=n, elems 0..7 = K0..7,  8..15 = K16..23
//   lane n+16       : N=n, elems 0..7 = K8..15, 8..15 = K24..31
// A-matrix (16x32 MxK) layout per ISA 7.12.2:
//   lane m    : elems 0..7 = K0..7,  8..15 = K16..23
//   lane m+16 : elems 0..7 = K8..15, 8..15 = K24..31

__device__ __forceinline__ float elu1(float x) {
  return x > 0.f ? x : __expf(x) - 1.f;
}

__device__ __forceinline__ void load_afrag(BFrag16& a, const __bf16* rowp) {
  a.h[0] = *(const v8bf*)(rowp);
  a.h[1] = *(const v8bf*)(rowp + 16);
}

template<int CHUNKS, int STRIDE>
__device__ __forceinline__ void conv_layer(
    const __bf16* __restrict__ bufIn, __bf16* __restrict__ bufOut,
    int Rin, int fragBase, int pl,
    const __bf16* __restrict__ wsB, const float* __restrict__ wsBias, int tid)
{
  const int wave = tid >> 5;
  const int lane = tid & 31;
  const int half = lane >> 4;
  const int r16  = lane & 15;
  const int Rout = Rin - 4;
  const int ntm  = (Rout + 15) >> 4;     // 8 or 9 -> tile0 always, tile1 sometimes

  const int m0 = wave << 4;
  const int m1 = (wave + 8) << 4;
  const bool has1 = (wave + 8) < ntm;

  v8f acc0[4] = {};
  v8f acc1[4] = {};
  const __bf16* bBase = wsB + (size_t)fragBase * 512 + lane * 16;
  // tile0 rows: m0 + r16 + tap <= 112+15+4 = 131 < 132 <= Rin  -> no clamp needed
  const __bf16* aBase0 = bufIn + (m0 + r16) * STRIDE + 8 * half;

#pragma unroll 1
  for (int tc = 0; tc < 5 * CHUNKS; ++tc) {
    const int tap = tc / CHUNKS;
    const int c   = tc % CHUNKS;
    // B fragments for this (tap, chunk): 4 N-tiles, live only this iteration
    const __bf16* bp = bBase + (size_t)tc * 4 * 512;
    v16bf b0 = *(const v16bf*)(bp);
    v16bf b1 = *(const v16bf*)(bp + 512);
    v16bf b2 = *(const v16bf*)(bp + 1024);
    v16bf b3 = *(const v16bf*)(bp + 1536);

    BFrag16 a0;
    load_afrag(a0, aBase0 + tap * STRIDE + c * 32);
    acc0[0] = __builtin_amdgcn_wmma_f32_16x16x32_bf16(false, a0.v, false, b0, (short)0, acc0[0], false, false);
    acc0[1] = __builtin_amdgcn_wmma_f32_16x16x32_bf16(false, a0.v, false, b1, (short)0, acc0[1], false, false);
    acc0[2] = __builtin_amdgcn_wmma_f32_16x16x32_bf16(false, a0.v, false, b2, (short)0, acc0[2], false, false);
    acc0[3] = __builtin_amdgcn_wmma_f32_16x16x32_bf16(false, a0.v, false, b3, (short)0, acc0[3], false, false);

    if (has1) {                          // wave-uniform branch
      int r1 = m1 + r16 + tap;
      r1 = r1 < Rin ? r1 : Rin - 1;      // clamped rows feed only never-read outputs
      BFrag16 a1;
      load_afrag(a1, bufIn + r1 * STRIDE + 8 * half + c * 32);
      acc1[0] = __builtin_amdgcn_wmma_f32_16x16x32_bf16(false, a1.v, false, b0, (short)0, acc1[0], false, false);
      acc1[1] = __builtin_amdgcn_wmma_f32_16x16x32_bf16(false, a1.v, false, b1, (short)0, acc1[1], false, false);
      acc1[2] = __builtin_amdgcn_wmma_f32_16x16x32_bf16(false, a1.v, false, b2, (short)0, acc1[2], false, false);
      acc1[3] = __builtin_amdgcn_wmma_f32_16x16x32_bf16(false, a1.v, false, b3, (short)0, acc1[3], false, false);
    }
  }

  // epilogue: bias + ELU -> bf16 LDS, unconditional stores (no EXEC churn).
  // C/D layout: elem r of lane l => M = m0 + r + 8*half, N = l%16.
  // Rows >= Rout (tile1 tail only) receive finite garbage that is never read:
  // max written row = 143 within both buffer capacities.
#pragma unroll
  for (int nt = 0; nt < 4; ++nt) {
    const float bv = wsBias[pl * 64 + nt * 16 + r16];
    __bf16* outp0 = bufOut + (m0 + 8 * half) * 64 + nt * 16 + r16;
#pragma unroll
    for (int r = 0; r < 8; ++r) {
      outp0[r * 64] = (__bf16)elu1(acc0[nt][r] + bv);
    }
    if (has1) {
      __bf16* outp1 = bufOut + (m1 + 8 * half) * 64 + nt * 16 + r16;
#pragma unroll
      for (int r = 0; r < 8; ++r) {
        outp1[r * 64] = (__bf16)elu1(acc1[nt][r] + bv);
      }
    }
  }
}

// ---------------------------------------------------------------------------
// Weight prep: fold BN into conv weights/bias, swizzle into B fragments (bf16)
// ---------------------------------------------------------------------------
__global__ __launch_bounds__(256) void prep_weights_kernel(
    const float* __restrict__ W, const float* __restrict__ bb,
    const float* __restrict__ beta, const float* __restrict__ gamma,
    const float* __restrict__ mean, const float* __restrict__ var,
    int cin, int cinChunks, int fragBase, int pl,
    __bf16* __restrict__ wsB, float* __restrict__ wsBias)
{
  const int f    = blockIdx.x;      // local fragment: ((tap*cinChunks + c)*4 + nt)
  const int nt   = f & 3;
  const int rest = f >> 2;
  const int c    = rest % cinChunks;
  const int tap  = rest / cinChunks;
  for (int idx = threadIdx.x; idx < 512; idx += 256) {
    const int lane = idx >> 4, e = idx & 15;
    const int K  = c * 32 + ((e < 8) ? 8 * (lane >> 4) + e : 16 + 8 * (lane >> 4) + (e - 8));
    const int co = nt * 16 + (lane & 15);
    float v = 0.f;
    if (K < cin && co < DIM) {
      const float scale = gamma[co] * rsqrtf(var[co] + BN_EPS);
      v = W[((size_t)tap * cin + K) * DIM + co] * scale;
    }
    wsB[(size_t)(fragBase + f) * 512 + idx] = (__bf16)v;
  }
  if (f == 0 && threadIdx.x < 64) {
    const int co = threadIdx.x;
    float bv = 0.f;
    if (co < DIM) {
      const float scale = gamma[co] * rsqrtf(var[co] + BN_EPS);
      bv = scale * (bb[co] - mean[co]) + beta[co];
    }
    wsBias[pl * 64 + co] = bv;
  }
}

// ---------------------------------------------------------------------------
// Stage 1: q = elu(bn(conv)) x5 over concat(de, p, dx) -> q1 (bf16, ch-padded)
// ---------------------------------------------------------------------------
__global__ __launch_bounds__(256) void stage1_kernel(
    const float* __restrict__ de, const float* __restrict__ p,
    const float* __restrict__ dx, __bf16* __restrict__ q1,
    const __bf16* __restrict__ wsB, const float* __restrict__ wsBias)
{
  __shared__ __align__(16) __bf16 smem[BUF0_ELEMS + BUF1_ELEMS];
  __bf16* buf0 = smem;
  __bf16* buf1 = smem + BUF0_ELEMS;
  const int tid = threadIdx.x;
  const int b = blockIdx.y;
  const int tileStart = blockIdx.x * TILE;

  if (tid < 16) buf0[R0 * S1 + tid] = (__bf16)0.f;   // zero overhang pad
  for (int i = tid; i < R0 * S1; i += 256) {
    const int row = i / S1, ch = i % S1;
    const int pos = tileStart - HALO + row;
    float v = 0.f;
    if (pos >= 0 && pos < SEQL && ch < 3 * DIM) {
      const size_t base = ((size_t)b * SEQL + pos) * DIM;
      v = ch < DIM ? de[base + ch]
        : ch < 2 * DIM ? p[base + ch - DIM]
        : dx[base + ch - 2 * DIM];
    }
    buf0[i] = (__bf16)v;
  }
  __syncthreads();
  conv_layer<5, S1>(buf0, buf1, 148,   0, 0, wsB, wsBias, tid); __syncthreads();
  conv_layer<2, 64>(buf1, buf0, 144, 100, 1, wsB, wsBias, tid); __syncthreads();
  conv_layer<2, 64>(buf0, buf1, 140, 140, 2, wsB, wsBias, tid); __syncthreads();
  conv_layer<2, 64>(buf1, buf0, 136, 180, 3, wsB, wsBias, tid); __syncthreads();
  conv_layer<2, 64>(buf0, buf1, 132, 220, 4, wsB, wsBias, tid); __syncthreads();
  for (int i = tid; i < TILE * 64; i += 256) {
    const int row = i >> 6, ch = i & 63;
    q1[((size_t)b * SEQL + tileStart + row) * 64 + ch] = buf1[row * 64 + ch];
  }
}

// ---------------------------------------------------------------------------
// Stage 2: input concat(dI, q1[perm], dx[perm]); scatter out[b, perm[b,j]] = q[b,j]
// ---------------------------------------------------------------------------
__global__ __launch_bounds__(256) void stage2_kernel(
    const float* __restrict__ dI, const float* __restrict__ dx,
    const __bf16* __restrict__ q1, const int* __restrict__ perm,
    float* __restrict__ out,
    const __bf16* __restrict__ wsB, const float* __restrict__ wsBias)
{
  __shared__ __align__(16) __bf16 smem[BUF0_ELEMS + BUF1_ELEMS];
  __bf16* buf0 = smem;
  __bf16* buf1 = smem + BUF0_ELEMS;
  const int tid = threadIdx.x;
  const int b = blockIdx.y;
  const int tileStart = blockIdx.x * TILE;

  if (tid < 16) buf0[R0 * S1 + tid] = (__bf16)0.f;
  for (int i = tid; i < R0 * S1; i += 256) {
    const int row = i / S1, ch = i % S1;
    const int pos = tileStart - HALO + row;
    float v = 0.f;
    if (pos >= 0 && pos < SEQL && ch < 3 * DIM) {
      if (ch < DIM) {
        v = dI[((size_t)b * SEQL + pos) * DIM + ch];
      } else {
        const int pp = perm[b * SEQL + pos];
        if (ch < 2 * DIM) v = (float)q1[((size_t)b * SEQL + pp) * 64 + (ch - DIM)];
        else              v = dx[((size_t)b * SEQL + pp) * DIM + (ch - 2 * DIM)];
      }
    }
    buf0[i] = (__bf16)v;
  }
  __syncthreads();
  conv_layer<5, S1>(buf0, buf1, 148, 260, 5, wsB, wsBias, tid); __syncthreads();
  conv_layer<2, 64>(buf1, buf0, 144, 360, 6, wsB, wsBias, tid); __syncthreads();
  conv_layer<2, 64>(buf0, buf1, 140, 400, 7, wsB, wsBias, tid); __syncthreads();
  conv_layer<2, 64>(buf1, buf0, 136, 440, 8, wsB, wsBias, tid); __syncthreads();
  conv_layer<2, 64>(buf0, buf1, 132, 480, 9, wsB, wsBias, tid); __syncthreads();
  // out[b, perm[b,j], :] = q[b, j, :]   (== permute by inv_perm)
  for (int i = tid; i < TILE * DIM; i += 256) {
    const int row = i / DIM, ch = i % DIM;
    const int pp = perm[b * SEQL + tileStart + row];
    out[((size_t)b * SEQL + pp) * DIM + ch] = (float)buf1[row * 64 + ch];
  }
}

// ---------------------------------------------------------------------------
// Host launch
// ---------------------------------------------------------------------------
extern "C" void kernel_launch(void* const* d_in, const int* in_sizes, int n_in,
                              void* d_out, int out_size, void* d_ws, size_t ws_size,
                              hipStream_t stream) {
  (void)n_in; (void)out_size; (void)ws_size;
  // Detect flattening order: JAX sorted pytree (dI,de,dx,inv_perm,p,params...,perm)
  // vs insertion order (dx,de,dI,p,params...,perm,inv_perm).
  const bool sortedOrder = (in_sizes[3] == BATCH * SEQL);

  const float *dx, *de, *dI, *p;
  const int* perm;
  int pbase, fBeta, fGamma;   // field index of beta/gamma within a unit (W=0,b=1,mean=4,var=5)
  if (sortedOrder) {
    dI = (const float*)d_in[0]; de = (const float*)d_in[1];
    dx = (const float*)d_in[2]; p  = (const float*)d_in[4];
    perm = (const int*)d_in[65];
    pbase = 5; fBeta = 2; fGamma = 3;          // sorted: W,b,beta,gamma,mean,var
  } else {
    dx = (const float*)d_in[0]; de = (const float*)d_in[1];
    dI = (const float*)d_in[2]; p  = (const float*)d_in[3];
    perm = (const int*)d_in[64];
    pbase = 4; fBeta = 3; fGamma = 2;          // insertion: W,b,gamma,beta,mean,var
  }

  // Workspace layout: [q1 bf16 256x2048x64][520 weight fragments x 1KB][bias f32 10x64]
  const size_t Q1_BYTES = (size_t)BATCH * SEQL * 64 * 2;
  __bf16* q1     = (__bf16*)d_ws;
  __bf16* wsB    = (__bf16*)((char*)d_ws + Q1_BYTES);
  float*  wsBias = (float*)((char*)d_ws + Q1_BYTES + (size_t)520 * 1024);

  static const int kCin[10]      = {150, 50, 50, 50, 50, 150, 50, 50, 50, 50};
  static const int kChunks[10]   = {5, 2, 2, 2, 2, 5, 2, 2, 2, 2};
  static const int kFragBase[10] = {0, 100, 140, 180, 220, 260, 360, 400, 440, 480};

  for (int pl = 0; pl < 10; ++pl) {
    // processing order: x1..x5 (stage1) then I1..I5 (stage2)
    int unit;
    if (sortedOrder) unit = (pl < 5) ? 5 + pl : pl - 5;  // units sorted I1..I5,x1..x5
    else             unit = pl;                           // units x1..x5,I1..I5
    const int u0 = pbase + unit * 6;
    prep_weights_kernel<<<kChunks[pl] * 20, 256, 0, stream>>>(
        (const float*)d_in[u0 + 0], (const float*)d_in[u0 + 1],
        (const float*)d_in[u0 + fBeta], (const float*)d_in[u0 + fGamma],
        (const float*)d_in[u0 + 4], (const float*)d_in[u0 + 5],
        kCin[pl], kChunks[pl], kFragBase[pl], pl, wsB, wsBias);
  }

  dim3 grid(SEQL / TILE, BATCH);
  stage1_kernel<<<grid, 256, 0, stream>>>(de, p, dx, q1, wsB, wsBias);
  stage2_kernel<<<grid, 256, 0, stream>>>(dI, dx, q1, perm, (float*)d_out, wsB, wsBias);
}
